// ActGCNLayer_21079699489214
// MI455X (gfx1250) — compile-verified, matrix operand order checked by reading, not run on hardware
//
#include <hip/hip_runtime.h>

#define DIN  128
#define DOUT 128
#define EPS  1e-5f

typedef __attribute__((ext_vector_type(8)))  __bf16 v8bf;
typedef __attribute__((ext_vector_type(16))) __bf16 v16bf;
typedef __attribute__((ext_vector_type(8)))  float  v8f;
typedef __attribute__((ext_vector_type(4)))  float  v4f;

// ---------------------------------------------------------------------------
// f32 -> bf16 (round to nearest even), elementwise
// ---------------------------------------------------------------------------
__global__ void cvt_bf16_kernel(const float* __restrict__ in,
                                unsigned short* __restrict__ out, int n) {
    int i = blockIdx.x * blockDim.x + threadIdx.x;
    if (i < n) {
        unsigned u = __float_as_uint(in[i]);
        unsigned r = (u + 0x7FFFu + ((u >> 16) & 1u)) >> 16;
        out[i] = (unsigned short)r;
    }
}

__global__ void zero_i32_kernel(int* __restrict__ p, int n) {
    int i = blockIdx.x * blockDim.x + threadIdx.x;
    if (i < n) p[i] = 0;
}

// ---------------------------------------------------------------------------
// Dual GEMM: support = x @ W^T + b ; res = x @ Wres^T + bres
// - Both weight matrices staged into LDS once per block via
//   global_load_async_to_lds_b128 (ASYNCcnt path), then B fragments come from
//   LDS (ds_load_b128) while A fragments stream from global.
// - One wave per 16-row M-tile; A fragment reused across 8 N-tiles and both
//   weight matrices.
// A (16x32 bf16) per-lane layout: lane L(0..15)=row L, K={klo..+7, klo+16..}
// with klo = (lane>=16 ? 8 : 0); built from two 16B row-major loads.
// B (32x16) = W^T column n = W row n -> identical addressing on W.
// ---------------------------------------------------------------------------
__global__ __launch_bounds__(256)
void gemm_dual_wmma_kernel(const __bf16* __restrict__ xb,
                           const __bf16* __restrict__ wb,
                           const __bf16* __restrict__ wrb,
                           const float* __restrict__ bias,
                           const float* __restrict__ biasr,
                           float* __restrict__ support,
                           float* __restrict__ res,
                           int nrows) {
    __shared__ __bf16 lds_wb[DOUT * DIN];   // 32 KB
    __shared__ __bf16 lds_wr[DOUT * DIN];   // 32 KB

    // ---- stage W and Wres into LDS with async global->LDS DMA ----
    {
        const int t = threadIdx.x;
#pragma unroll
        for (int i = 0; i < 8; ++i) {
            const int byteoff = i * 4096 + t * 16;   // 256 thr * 16B * 8 = 32KB
            unsigned l0 = (unsigned)(unsigned long long)
                              (const void*)((const char*)lds_wb + byteoff);
            const void* g0 = (const char*)wb + byteoff;
            asm volatile("global_load_async_to_lds_b128 %0, %1, off"
                         :: "v"(l0), "v"(g0) : "memory");
            unsigned l1 = (unsigned)(unsigned long long)
                              (const void*)((const char*)lds_wr + byteoff);
            const void* g1 = (const char*)wrb + byteoff;
            asm volatile("global_load_async_to_lds_b128 %0, %1, off"
                         :: "v"(l1), "v"(g1) : "memory");
        }
        asm volatile("s_wait_asynccnt 0" ::: "memory");
        __syncthreads();
    }

    const int lane  = threadIdx.x & 31;
    const int wave  = threadIdx.x >> 5;
    const int mtile = blockIdx.x * 8 + wave;
    if (mtile * 16 >= nrows) return;                  // wave-uniform exit

    const int l15  = lane & 15;
    const int koff = (lane >> 4) ? 8 : 0;
    const int arow = mtile * 16 + l15;

    v16bf a[4];
#pragma unroll
    for (int kt = 0; kt < 4; ++kt) {
        const __bf16* p = xb + arow * DIN + kt * 32 + koff;
        v8bf lo = *(const v8bf*)(p);
        v8bf hi = *(const v8bf*)(p + 16);
        a[kt] = __builtin_shufflevector(lo, hi, 0,1,2,3,4,5,6,7,
                                                8,9,10,11,12,13,14,15);
    }

    const int srow0 = mtile * 16 + (koff ? 8 : 0);    // C/D: vgpr v -> row v(+8)
#pragma unroll
    for (int nt = 0; nt < 8; ++nt) {
        const int col = nt * 16 + l15;
        const float bs = bias[col];
        const float br = biasr[col];
        v8f accS = { bs, bs, bs, bs, bs, bs, bs, bs };
        v8f accR = { br, br, br, br, br, br, br, br };
#pragma unroll
        for (int kt = 0; kt < 4; ++kt) {
            const __bf16* pw = lds_wb + col * DIN + kt * 32 + koff;
            const __bf16* pr = lds_wr + col * DIN + kt * 32 + koff;
            v8bf wlo = *(const v8bf*)(pw);
            v8bf whi = *(const v8bf*)(pw + 16);
            v8bf rlo = *(const v8bf*)(pr);
            v8bf rhi = *(const v8bf*)(pr + 16);
            v16bf bw = __builtin_shufflevector(wlo, whi, 0,1,2,3,4,5,6,7,
                                                         8,9,10,11,12,13,14,15);
            v16bf bv = __builtin_shufflevector(rlo, rhi, 0,1,2,3,4,5,6,7,
                                                         8,9,10,11,12,13,14,15);
            accS = __builtin_amdgcn_wmma_f32_16x16x32_bf16(
                       false, a[kt], false, bw, (short)0, accS, false, false);
            accR = __builtin_amdgcn_wmma_f32_16x16x32_bf16(
                       false, a[kt], false, bv, (short)0, accR, false, false);
        }
#pragma unroll
        for (int v = 0; v < 8; ++v) {
            support[(srow0 + v) * DOUT + col] = accS[v];
            res[(srow0 + v) * DOUT + col]     = accR[v];
        }
    }
}

// ---------------------------------------------------------------------------
// CSR build: degree histogram -> exclusive scan -> edge scatter
// ---------------------------------------------------------------------------
__global__ void degree_kernel(const int* __restrict__ dst, int* __restrict__ deg,
                              int e) {
    int i = blockIdx.x * blockDim.x + threadIdx.x;
    if (i < e) atomicAdd(&deg[dst[i]], 1);
}

__global__ __launch_bounds__(1024)
void scan_kernel(const int* __restrict__ deg, int* __restrict__ rowptr,
                 int* __restrict__ cursor, int n) {
    __shared__ int sh[1024];
    const int t = threadIdx.x;
    int carry = 0;
    for (int base = 0; base < n; base += 1024) {
        const int i = base + t;
        const int v = (i < n) ? deg[i] : 0;
        sh[t] = v;
        __syncthreads();
        int x = v;
        for (int off = 1; off < 1024; off <<= 1) {
            int y = (t >= off) ? sh[t - off] : 0;
            __syncthreads();
            x += y;
            sh[t] = x;
            __syncthreads();
        }
        if (i < n) {
            const int excl = carry + x - v;
            rowptr[i] = excl;
            cursor[i] = excl;
        }
        const int tot = sh[1023];
        __syncthreads();
        carry += tot;
    }
    if (t == 0) rowptr[n] = carry;
}

__global__ void scatter_kernel(const int* __restrict__ src,
                               const int* __restrict__ dst,
                               const float* __restrict__ ew,
                               int* __restrict__ cursor,
                               int* __restrict__ csr_src,
                               float* __restrict__ csr_w, int e) {
    int i = blockIdx.x * blockDim.x + threadIdx.x;
    if (i < e) {
        int p = atomicAdd(&cursor[dst[i]], 1);
        csr_src[p] = src[i];
        csr_w[p]   = ew[i];
    }
}

// ---------------------------------------------------------------------------
// Gather SpMM: one wave per node, lane owns 4 consecutive features (float4).
// support is L2-resident (25.6 MB << 192 MB L2) -> pure streaming L2 reads.
// ---------------------------------------------------------------------------
__global__ __launch_bounds__(256)
void spmm_gather_kernel(const int* __restrict__ rowptr,
                        const int* __restrict__ csr_src,
                        const float* __restrict__ csr_w,
                        const float* __restrict__ support,
                        float* __restrict__ out, int n) {
    const int node = (int)((blockIdx.x * blockDim.x + threadIdx.x) >> 5);
    const int lane = threadIdx.x & 31;
    if (node >= n) return;
    const int beg = rowptr[node];
    const int end = rowptr[node + 1];
    const v4f* sup4 = (const v4f*)support;
    v4f acc = { 0.f, 0.f, 0.f, 0.f };
    for (int e = beg; e < end; ++e) {
        const int s   = csr_src[e];
        const float w = csr_w[e];
        if (e + 1 < end)
            __builtin_prefetch(&sup4[csr_src[e + 1] * 32 + lane], 0, 1);
        v4f v = sup4[s * 32 + lane];
        acc = acc + v * w;
    }
    ((v4f*)out)[node * 32 + lane] = acc;
}

// ---------------------------------------------------------------------------
// Column-wise sum / sum-of-squares (for batch-norm statistics)
// ---------------------------------------------------------------------------
__global__ __launch_bounds__(256)
void colreduce_kernel(const float* __restrict__ out, float* __restrict__ gsum,
                      float* __restrict__ gsqs, int nrows) {
    __shared__ float shS[128];
    __shared__ float shQ[128];
    const int col  = threadIdx.x & 127;
    const int half = threadIdx.x >> 7;
    const int rpb  = (nrows + (int)gridDim.x - 1) / (int)gridDim.x;
    const int r0   = (int)blockIdx.x * rpb;
    int r1 = r0 + rpb; if (r1 > nrows) r1 = nrows;
    float s = 0.f, q = 0.f;
    for (int r = r0 + half; r < r1; r += 2) {
        float v = out[r * DOUT + col];
        s += v;
        q += v * v;
    }
    if (half) { shS[col] = s; shQ[col] = q; }
    __syncthreads();
    if (!half) {
        s += shS[col];
        q += shQ[col];
        atomicAdd(&gsum[col], s);
        atomicAdd(&gsqs[col], q);
    }
}

// ---------------------------------------------------------------------------
// Finalize: batchnorm (biased var) + affine + ReLU + residual add
// ---------------------------------------------------------------------------
__global__ void finalize_kernel(const float* __restrict__ out,
                                const float* __restrict__ res,
                                const float* __restrict__ gsum,
                                const float* __restrict__ gsqs,
                                const float* __restrict__ gamma,
                                const float* __restrict__ beta,
                                float* __restrict__ y, int total, float invN) {
    int i = blockIdx.x * blockDim.x + threadIdx.x;
    if (i >= total) return;
    const int col  = i & (DOUT - 1);
    const float mu = gsum[col] * invN;
    const float va = gsqs[col] * invN - mu * mu;
    const float rs = rsqrtf(va + EPS);
    float v = (out[i] - mu) * rs * gamma[col] + beta[col];
    v = fmaxf(v, 0.f) + res[i];
    y[i] = v;
}

// ---------------------------------------------------------------------------
extern "C" void kernel_launch(void* const* d_in, const int* in_sizes, int n_in,
                              void* d_out, int out_size, void* d_ws, size_t ws_size,
                              hipStream_t stream) {
    const float* x     = (const float*)d_in[0];
    const int*   eidx  = (const int*)d_in[1];
    const float* ew    = (const float*)d_in[2];
    const float* W     = (const float*)d_in[3];
    const float* b     = (const float*)d_in[4];
    const float* Wres  = (const float*)d_in[5];
    const float* bres  = (const float*)d_in[6];
    const float* gamma = (const float*)d_in[7];
    const float* beta  = (const float*)d_in[8];

    const int N = in_sizes[0] / DIN;   // 50000
    const int E = in_sizes[2];         // 1600000
    const int* srcp = eidx;
    const int* dstp = eidx + E;

    // ---- workspace carve-up (256B aligned) ----
    char* wp = (char*)d_ws;
    auto alloc = [&](size_t bytes) {
        void* p = (void*)wp;
        wp += (bytes + 255) & ~(size_t)255;
        return p;
    };
    unsigned short* xb      = (unsigned short*)alloc((size_t)N * DIN * 2);
    unsigned short* wb      = (unsigned short*)alloc((size_t)DOUT * DIN * 2);
    unsigned short* wrb     = (unsigned short*)alloc((size_t)DOUT * DIN * 2);
    float*          support = (float*)alloc((size_t)N * DOUT * 4);
    float*          res     = (float*)alloc((size_t)N * DOUT * 4);
    float*          outbuf  = (float*)alloc((size_t)N * DOUT * 4);
    int*            deg     = (int*)alloc((size_t)N * 4);
    int*            rowptr  = (int*)alloc((size_t)(N + 1) * 4);
    int*            cursor  = (int*)alloc((size_t)N * 4);
    int*            csr_src = (int*)alloc((size_t)E * 4);
    float*          csr_w   = (float*)alloc((size_t)E * 4);
    float*          gsum    = (float*)alloc(DOUT * 4);
    float*          gsqs    = (float*)alloc(DOUT * 4);

    const int T = 256;
    // zero accumulators (d_ws is poisoned, not re-zeroed between replays)
    zero_i32_kernel<<<(N + T - 1) / T, T, 0, stream>>>(deg, N);
    zero_i32_kernel<<<1, T, 0, stream>>>((int*)gsum, DOUT);
    zero_i32_kernel<<<1, T, 0, stream>>>((int*)gsqs, DOUT);

    // bf16 conversions
    cvt_bf16_kernel<<<((size_t)N * DIN + T - 1) / T, T, 0, stream>>>(x, xb, N * DIN);
    cvt_bf16_kernel<<<(DOUT * DIN + T - 1) / T, T, 0, stream>>>(W, wb, DOUT * DIN);
    cvt_bf16_kernel<<<(DOUT * DIN + T - 1) / T, T, 0, stream>>>(Wres, wrb, DOUT * DIN);

    // dual WMMA GEMM (weights staged in LDS via async global->LDS copies)
    const int mtiles = N / 16;
    gemm_dual_wmma_kernel<<<(mtiles + 7) / 8, 256, 0, stream>>>(
        (const __bf16*)xb, (const __bf16*)wb, (const __bf16*)wrb,
        b, bres, support, res, N);

    // CSR build
    degree_kernel<<<(E + T - 1) / T, T, 0, stream>>>(dstp, deg, E);
    scan_kernel<<<1, 1024, 0, stream>>>(deg, rowptr, cursor, N);
    scatter_kernel<<<(E + T - 1) / T, T, 0, stream>>>(srcp, dstp, ew, cursor,
                                                      csr_src, csr_w, E);

    // gather SpMM (one wave per node)
    spmm_gather_kernel<<<((size_t)N * 32 + T - 1) / T, T, 0, stream>>>(
        rowptr, csr_src, csr_w, support, outbuf, N);

    // batchnorm statistics
    colreduce_kernel<<<256, 256, 0, stream>>>(outbuf, gsum, gsqs, N);

    // normalize + relu + residual
    finalize_kernel<<<((size_t)N * DOUT + T - 1) / T, T, 0, stream>>>(
        outbuf, res, gsum, gsqs, gamma, beta, (float*)d_out, N * DOUT,
        1.0f / (float)N);
}
